// CausalWanSelfAttention_90744069029937
// MI455X (gfx1250) — matrix operand order repaired and checked by reading, not compiled, not run on hardware
//
#include <hip/hip_runtime.h>
#include <hip/hip_bf16.h>

// ---------------- compile-time problem constants (match reference) ----------
#define DIM      1536
#define NHEADS   12
#define HEAD_DIM 128
#define F_       3
#define H_       30
#define W_       52
#define L_       (F_ * H_ * W_)   // 4680
#define FRAME    (H_ * W_)        // 1560
#define LP       4736             // 74 * 64 (L padded to 64)
#define QT       (LP / 16)        // 296 query tiles
#define SINK     1
#define LOCAL    2

typedef __bf16 bf16_t;
typedef __attribute__((ext_vector_type(16))) __bf16 v16bf;
typedef __attribute__((ext_vector_type(8)))  __bf16 v8bf;
typedef __attribute__((ext_vector_type(8)))  float  v8f;
typedef __attribute__((ext_vector_type(4)))  int    v4i;

#define AS1 __attribute__((address_space(1)))
#define AS3 __attribute__((address_space(3)))

#if __has_builtin(__builtin_amdgcn_global_load_async_to_lds_b128)
#define HAVE_ASYNC_LDS 1
#endif

static __device__ __forceinline__ void cp_async16(const bf16_t* g, bf16_t* l) {
#if defined(HAVE_ASYNC_LDS)
  __builtin_amdgcn_global_load_async_to_lds_b128((AS1 v4i*)g, (AS3 v4i*)l, 0, 0);
#else
  *(v8bf*)l = *(const v8bf*)g;   // synchronous fallback
#endif
}

static __device__ __forceinline__ void wait_async_2() {
#if defined(HAVE_ASYNC_LDS)
#if __has_builtin(__builtin_amdgcn_s_wait_asynccnt)
  __builtin_amdgcn_s_wait_asynccnt(2);
#else
  asm volatile("s_wait_asynccnt 0x2" ::: "memory");
#endif
#endif
}
static __device__ __forceinline__ void wait_async_0() {
#if defined(HAVE_ASYNC_LDS)
#if __has_builtin(__builtin_amdgcn_s_wait_asynccnt)
  __builtin_amdgcn_s_wait_asynccnt(0);
#else
  asm volatile("s_wait_asynccnt 0x0" ::: "memory");
#endif
#endif
}

static __device__ __forceinline__ v16bf cat8(v8bf lo, v8bf hi) {
  v16bf r;
#pragma unroll
  for (int i = 0; i < 8; ++i) { r[i] = lo[i]; r[i + 8] = hi[i]; }
  return r;
}

// A-matrix 16x32 (MxK) bf16, contiguous k order (same as the B tiles):
// lanes 0-15 : row m=lane, K = k0+0..15; lanes 16-31: K = k0+16..31 (contiguous)
static __device__ __forceinline__ v16bf load_a_tile(const bf16_t* __restrict__ p,
                                                    int ld, int row_base, int k0, int lane) {
  int m    = lane & 15;
  int koff = (lane >> 4) ? 16 : 0;
  const v8bf* r = (const v8bf*)(p + (size_t)(row_base + m) * ld + k0 + koff);
  return cat8(r[0], r[1]);
}

// B-matrix 32x16 (KxN) bf16, source row-major src[n][k]:
// lanes 0-15: col n=lane, K=k0+0..15; lanes 16-31: K=k0+16..31 (contiguous 32B)
static __device__ __forceinline__ v16bf load_b_tile(const bf16_t* __restrict__ p,
                                                    int ld, int col_base, int k0, int lane) {
  int n    = lane & 15;
  int koff = (lane >> 4) ? 16 : 0;
  const v8bf* r = (const v8bf*)(p + (size_t)(col_base + n) * ld + k0 + koff);
  return cat8(r[0], r[1]);
}

static __device__ __forceinline__ v8f wmma_bf16(v16bf a, v16bf b, v8f c) {
  return __builtin_amdgcn_wmma_f32_16x16x32_bf16(false, a, false, b, (short)0, c, false, false);
}

// ---------------- conversion kernels ----------------------------------------
__global__ void cvt_x_kernel(const float* __restrict__ x, bf16_t* __restrict__ xb) {
  int i   = blockIdx.x * 256 + threadIdx.x;      // 0 .. LP*DIM-1
  int row = i / DIM;
  xb[i] = (row < L_) ? (bf16_t)x[i] : (bf16_t)0.0f;
}

__global__ void cvt_w_kernel(const float* __restrict__ w, bf16_t* __restrict__ wb) {
  int i = blockIdx.x * 256 + threadIdx.x;
  wb[i] = (bf16_t)w[i];
}

// ---------------- WMMA GEMM: 64x64 block tile, 4 waves, LDS-staged B --------
// B slice (32K x 64N, 4KB) is async-copied to LDS once per block per K-step,
// double-buffered; each wave computes a 16x64 strip.
// mode 0: fp32 row-major out, row-guarded          (final projection -> d_out)
// mode 1: bf16 row-major out [row][col]            (q/k pre-norm)
// mode 2: bf16 transposed out [col][row] (ld = LP) (v -> vT)
__global__ __launch_bounds__(128)
void gemm64x64(const bf16_t* __restrict__ A, const bf16_t* __restrict__ Wt,
               const float* __restrict__ bias, void* __restrict__ out,
               int guard_rows, int mode) {
  __shared__ bf16_t Bs[2][64 * 32];   // 2 x 4KB double buffer

  int tid      = threadIdx.x;
  int wave     = tid >> 5;
  int lane     = tid & 31;
  int row_base = blockIdx.x * 64 + wave * 16;
  int col_base = blockIdx.y * 64;

  // stage 64 rows x 32 bf16 = 256 x 16B chunks; each thread copies 2 chunks
  auto stage = [&](int buf, int k0) {
#pragma unroll
    for (int it = 0; it < 2; ++it) {
      int c   = it * 128 + tid;   // 0..255
      int r   = c >> 2;           // row within 64
      int off = (c & 3) * 8;      // bf16 offset within row
      cp_async16(Wt + (size_t)(col_base + r) * DIM + k0 + off,
                 &Bs[buf][r * 32 + off]);
    }
  };

  v8f acc[4];
#pragma unroll
  for (int t = 0; t < 4; ++t)
#pragma unroll
    for (int e = 0; e < 8; ++e) acc[t][e] = 0.0f;

  stage(0, 0);
  int buf = 0;
  for (int k0 = 0; k0 < DIM; k0 += 32) {
    if (k0 + 32 < DIM) { stage(buf ^ 1, k0 + 32); wait_async_2(); }
    else               { wait_async_0(); }
    __syncthreads();                         // buf fully staged for all waves

    v16bf a  = load_a_tile(A, DIM, row_base, k0, lane);
    int koff = (lane >> 4) ? 16 : 0;
#pragma unroll
    for (int t = 0; t < 4; ++t) {
      const v8bf* bp = (const v8bf*)&Bs[buf][(t * 16 + (lane & 15)) * 32 + koff];
      acc[t] = wmma_bf16(a, cat8(bp[0], bp[1]), acc[t]);
    }
    __syncthreads();                         // all reads done before overwrite
    buf ^= 1;
  }

  int half = lane >> 4;
  int n    = lane & 15;
#pragma unroll
  for (int t = 0; t < 4; ++t) {
    int   col = col_base + t * 16 + n;
    float bv  = bias[col];
#pragma unroll
    for (int e = 0; e < 8; ++e) {
      int   row = row_base + e + half * 8;
      float v   = acc[t][e] + bv;
      if (mode == 0) {
        if (row < guard_rows) ((float*)out)[(size_t)row * DIM + col] = v;
      } else if (mode == 1) {
        ((bf16_t*)out)[(size_t)row * DIM + col] = (bf16_t)v;
      } else {
        ((bf16_t*)out)[(size_t)col * LP + row] = (bf16_t)v;   // transposed V
      }
    }
  }
}

// ---------------- RMSNorm (full 1536) + 3-axis RoPE -> bf16 -----------------
__global__ __launch_bounds__(256)
void norm_rope_kernel(const bf16_t* __restrict__ qb, const bf16_t* __restrict__ kb,
                      const float* __restrict__ gq, const float* __restrict__ gk,
                      const float* __restrict__ freqs,
                      bf16_t* __restrict__ qr, bf16_t* __restrict__ kr) {
  int row = blockIdx.x;
  int tid = threadIdx.x;
  if (row >= L_) {   // zero pad rows so attention A-tiles stay finite
    for (int i = tid; i < DIM; i += 256) {
      qr[(size_t)row * DIM + i] = (bf16_t)0.0f;
      kr[(size_t)row * DIM + i] = (bf16_t)0.0f;
    }
    return;
  }
  const bf16_t* qrow = qb + (size_t)row * DIM;
  const bf16_t* krow = kb + (size_t)row * DIM;

  __shared__ float sq[256], sk[256];
  float aq = 0.0f, ak = 0.0f;
  for (int i = tid; i < DIM; i += 256) {
    float q = (float)qrow[i], k = (float)krow[i];
    aq += q * q; ak += k * k;
  }
  sq[tid] = aq; sk[tid] = ak;
  __syncthreads();
  for (int s = 128; s > 0; s >>= 1) {
    if (tid < s) { sq[tid] += sq[tid + s]; sk[tid] += sk[tid + s]; }
    __syncthreads();
  }
  float rq = rsqrtf(sq[0] * (1.0f / DIM) + 1e-6f);
  float rk = rsqrtf(sk[0] * (1.0f / DIM) + 1e-6f);

  int t_idx = row / FRAME;
  int rem   = row % FRAME;
  int h_idx = rem / W_;
  int w_idx = rem % W_;

  for (int p = tid; p < DIM / 2; p += 256) {   // pair index; element base = 2p
    int j   = p & 63;                          // pair index within head (c=64)
    int pos = (j < 22) ? t_idx : ((j < 43) ? h_idx : w_idx);
    float cs = freqs[pos * 128 + 2 * j];
    float sn = freqs[pos * 128 + 2 * j + 1];
    size_t i0 = (size_t)row * DIM + 2 * p;

    float a = (float)qrow[2 * p]     * rq * gq[2 * p];
    float b = (float)qrow[2 * p + 1] * rq * gq[2 * p + 1];
    qr[i0]     = (bf16_t)(a * cs - b * sn);
    qr[i0 + 1] = (bf16_t)(a * sn + b * cs);

    a = (float)krow[2 * p]     * rk * gk[2 * p];
    b = (float)krow[2 * p + 1] * rk * gk[2 * p + 1];
    kr[i0]     = (bf16_t)(a * cs - b * sn);
    kr[i0 + 1] = (bf16_t)(a * sn + b * cs);
  }
}

// ---------------- flash attention: one wave = (head, 16-query tile) ---------
__global__ __launch_bounds__(32)
void attn_kernel(const bf16_t* __restrict__ qr, const bf16_t* __restrict__ kr,
                 const bf16_t* __restrict__ vT, bf16_t* __restrict__ ao) {
  __shared__ float sP[16][33];   // P transpose staging (C-layout -> A-layout)

  int lane  = threadIdx.x;
  int half  = lane >> 4;
  int ln    = lane & 15;
  int h     = blockIdx.y;
  int qbase = blockIdx.x * 16;
  int hd0   = h * HEAD_DIM;

  v16bf qA[4];
#pragma unroll
  for (int c = 0; c < 4; ++c) qA[c] = load_a_tile(qr, DIM, qbase, hd0 + c * 32, lane);

  int last_row = qbase + 15 < L_ - 1 ? qbase + 15 : L_ - 1;
  int fq_max   = last_row / FRAME;
  int kend     = (fq_max + 1) * FRAME; if (kend > L_) kend = L_;
  int kend32   = (kend + 31) & ~31;

  v8f acc[4], m_run, l_run;
#pragma unroll
  for (int e = 0; e < 8; ++e) { m_run[e] = -1e30f; l_run[e] = 0.0f; }
#pragma unroll
  for (int t = 0; t < 4; ++t)
#pragma unroll
    for (int e = 0; e < 8; ++e) acc[t][e] = 0.0f;

  const float scale = 0.08838834764831845f;   // 1/sqrt(128)

  for (int kb = 0; kb < kend32; kb += 32) {
    // prefetch next chunk's K rows and V columns while we crunch this one
    if (kb + 32 < kend32) {
      __builtin_prefetch(kr + (size_t)(kb + 32 + lane) * DIM + hd0, 0, 1);
      __builtin_prefetch(vT + (size_t)(hd0 + lane) * LP + kb + 32, 0, 1);
      __builtin_prefetch(vT + (size_t)(hd0 + 32 + lane) * LP + kb + 32, 0, 1);
      __builtin_prefetch(vT + (size_t)(hd0 + 64 + lane) * LP + kb + 32, 0, 1);
      __builtin_prefetch(vT + (size_t)(hd0 + 96 + lane) * LP + kb + 32, 0, 1);
    }

    v8f s0, s1;
#pragma unroll
    for (int e = 0; e < 8; ++e) { s0[e] = 0.0f; s1[e] = 0.0f; }
#pragma unroll
    for (int c = 0; c < 4; ++c) {
      v16bf b0 = load_b_tile(kr, DIM, kb,      hd0 + c * 32, lane);
      v16bf b1 = load_b_tile(kr, DIM, kb + 16, hd0 + c * 32, lane);
      s0 = wmma_bf16(qA[c], b0, s0);
      s1 = wmma_bf16(qA[c], b1, s1);
    }

    int  n0  = kb + ln, n1 = n0 + 16;
    int  fk0 = n0 / FRAME, fk1 = n1 / FRAME;
    bool v0  = n0 < L_,    v1  = n1 < L_;

#pragma unroll
    for (int e = 0; e < 8; ++e) {
      int mrow = qbase + e + half * 8;
      int fq   = mrow / FRAME;
      bool a0 = v0 && (fk0 <= fq) && ((fk0 < SINK) || (fq - fk0 < LOCAL));
      bool a1 = v1 && (fk1 <= fq) && ((fk1 < SINK) || (fq - fk1 < LOCAL));
      float x0 = a0 ? s0[e] * scale : -1e30f;
      float x1 = a1 ? s1[e] * scale : -1e30f;

      float mx = fmaxf(x0, x1);
#pragma unroll
      for (int off = 1; off < 16; off <<= 1) mx = fmaxf(mx, __shfl_xor(mx, off, 32));
      float mnew  = fmaxf(m_run[e], mx);
      float alpha = __expf(m_run[e] - mnew);
      float p0    = __expf(x0 - mnew);
      float p1    = __expf(x1 - mnew);
      float rs    = p0 + p1;
#pragma unroll
      for (int off = 1; off < 16; off <<= 1) rs += __shfl_xor(rs, off, 32);
      l_run[e] = l_run[e] * alpha + rs;
      m_run[e] = mnew;
      sP[e + half * 8][ln]      = p0;
      sP[e + half * 8][ln + 16] = p1;
#pragma unroll
      for (int t = 0; t < 4; ++t) acc[t][e] *= alpha;
    }
    __syncthreads();

    v16bf pA;   // A-layout 16x32 of P
    {
      int koff = half * 16;
#pragma unroll
      for (int j = 0; j < 8; ++j) {
        pA[j]     = (bf16_t)sP[ln][koff + j];
        pA[j + 8] = (bf16_t)sP[ln][koff + 8 + j];
      }
    }
    __syncthreads();

#pragma unroll
    for (int t = 0; t < 4; ++t) {
      // V B-tile: column n = d-index (vT row), K = tokens kb..kb+31 (contiguous)
      const v8bf* r = (const v8bf*)(vT + (size_t)(hd0 + t * 16 + ln) * LP + kb + half * 16);
      v16bf vB = cat8(r[0], r[1]);
      acc[t] = wmma_bf16(pA, vB, acc[t]);
    }
  }

#pragma unroll
  for (int t = 0; t < 4; ++t) {
    int col = hd0 + t * 16 + ln;
#pragma unroll
    for (int e = 0; e < 8; ++e) {
      int row = qbase + e + half * 8;
      ao[(size_t)row * DIM + col] = (bf16_t)(acc[t][e] / l_run[e]);
    }
  }
}

// ---------------- host side --------------------------------------------------
extern "C" void kernel_launch(void* const* d_in, const int* in_sizes, int n_in,
                              void* d_out, int out_size, void* d_ws, size_t ws_size,
                              hipStream_t stream) {
  const float* x     = (const float*)d_in[0];
  const float* wq    = (const float*)d_in[1];
  const float* bq    = (const float*)d_in[2];
  const float* wk    = (const float*)d_in[3];
  const float* bk    = (const float*)d_in[4];
  const float* wv    = (const float*)d_in[5];
  const float* bv    = (const float*)d_in[6];
  const float* wo    = (const float*)d_in[7];
  const float* bo    = (const float*)d_in[8];
  const float* gq    = (const float*)d_in[9];
  const float* gk    = (const float*)d_in[10];
  const float* freqs = (const float*)d_in[11];
  // d_in[12..] : seq_lens, grid_sizes, F, H, W, sink, local — fixed at compile time.

  char*  ws  = (char*)d_ws;
  size_t off = 0;
  auto alloc = [&](size_t bytes) -> void* {
    void* p = ws + off;
    off = (off + bytes + 255) & ~(size_t)255;
    return p;
  };
  bf16_t* xb  = (bf16_t*)alloc((size_t)LP * DIM * 2);
  bf16_t* wqb = (bf16_t*)alloc((size_t)DIM * DIM * 2);
  bf16_t* wkb = (bf16_t*)alloc((size_t)DIM * DIM * 2);
  bf16_t* wvb = (bf16_t*)alloc((size_t)DIM * DIM * 2);
  bf16_t* wob = (bf16_t*)alloc((size_t)DIM * DIM * 2);
  bf16_t* qpre= (bf16_t*)alloc((size_t)LP * DIM * 2);
  bf16_t* kpre= (bf16_t*)alloc((size_t)LP * DIM * 2);
  bf16_t* qro = (bf16_t*)alloc((size_t)LP * DIM * 2);
  bf16_t* kro = (bf16_t*)alloc((size_t)LP * DIM * 2);
  bf16_t* vT  = (bf16_t*)alloc((size_t)DIM * LP * 2);
  bf16_t* ao  = (bf16_t*)alloc((size_t)LP * DIM * 2);

  cvt_x_kernel<<<(LP * DIM) / 256, 256, 0, stream>>>(x, xb);
  cvt_w_kernel<<<(DIM * DIM) / 256, 256, 0, stream>>>(wq, wqb);
  cvt_w_kernel<<<(DIM * DIM) / 256, 256, 0, stream>>>(wk, wkb);
  cvt_w_kernel<<<(DIM * DIM) / 256, 256, 0, stream>>>(wv, wvb);
  cvt_w_kernel<<<(DIM * DIM) / 256, 256, 0, stream>>>(wo, wob);

  dim3 ggrid(LP / 64, DIM / 64);
  gemm64x64<<<ggrid, 128, 0, stream>>>(xb, wqb, bq, qpre, LP, 1);
  gemm64x64<<<ggrid, 128, 0, stream>>>(xb, wkb, bk, kpre, LP, 1);
  gemm64x64<<<ggrid, 128, 0, stream>>>(xb, wvb, bv, vT,   LP, 2);

  norm_rope_kernel<<<LP, 256, 0, stream>>>(qpre, kpre, gq, gk, freqs, qro, kro);

  attn_kernel<<<dim3(QT, NHEADS), 32, 0, stream>>>(qro, kro, vT, ao);

  gemm64x64<<<ggrid, 128, 0, stream>>>(ao, wob, bo, d_out, L_, 0);
}
